// CausalSelfAttention_54769422958705
// MI455X (gfx1250) — matrix operand order repaired and checked by reading, not compiled, not run on hardware
//
#include <hip/hip_runtime.h>
#include <hip/hip_bf16.h>
#include <cstdint>

// ---------------------------------------------------------------------------
// CausalSelfAttention (GQA + RoPE) for gfx1250 / MI455X.
// - All matmuls on v_wmma_f32_16x16x32_f16 (f16 in, f32 accumulate).
// - Flash-attention streaming (TxT scores never touch HBM).
// - K-tiles / f16 A-tiles staged with global_load_async_to_lds_b128 (ASYNCcnt).
// ---------------------------------------------------------------------------

typedef __attribute__((ext_vector_type(16))) _Float16 v16h;
typedef __attribute__((ext_vector_type(8)))  _Float16 v8h;
typedef __attribute__((ext_vector_type(2)))  _Float16 v2h;
typedef __attribute__((ext_vector_type(8)))  float    v8f;

#define N_HEAD   16
#define N_KVH    4
#define HEAD_D   128
#define SEQ_T    2048
#define BATCH    2
#define C_DIM    (N_HEAD * HEAD_D)   // 2048
#define CKV_DIM  (N_KVH  * HEAD_D)   // 512
#define M_ROWS   (BATCH * SEQ_T)     // 4096

__device__ __forceinline__ v8f wmma_f16(v16h a, v16h b, v8f c) {
  return __builtin_amdgcn_wmma_f32_16x16x32_f16(
      /*neg_a=*/false, a, /*neg_b=*/false, b,
      /*c_mod=*/(short)0, c, /*reuse_a=*/false, /*reuse_b=*/false);
}

// Async copy of 16B from global to LDS (per lane). Tracked by ASYNCcnt.
__device__ __forceinline__ void async_copy_b128(unsigned lds_off,
                                                const void* gaddr) {
  asm volatile("global_load_async_to_lds_b128 %0, %1, off"
               :
               : "v"(lds_off), "v"((uint64_t)gaddr)
               : "memory");
}
__device__ __forceinline__ void wait_async0() {
  asm volatile("s_wait_asynccnt 0x0" ::: "memory");
}
// Low 32 bits of a generic shared pointer = LDS byte offset.
__device__ __forceinline__ unsigned lds_off32(const void* p) {
  return (unsigned)(uintptr_t)p;
}

// A-operand fragment (16x32 f16): lane half lh holds contiguous K = lh*16..+16.
// p must point at (row_base + lh*8), 16B aligned.
__device__ __forceinline__ v16h load_a_frag(const _Float16* p) {
  const int lhh = (threadIdx.x & 31) >> 4;
  v8h lo = *(const v8h*)(p + lhh * 8);
  v8h hi = *(const v8h*)(p + lhh * 8 + 8);
  v16h r;
#pragma unroll
  for (int i = 0; i < 8; ++i) { r[i] = lo[i]; r[i + 8] = hi[i]; }
  return r;
}

// B-operand fragment (32x16 f16): lane half lh holds contiguous K = lh*16..+16.
// p must point at (col_base + lh*16), 32B aligned.
__device__ __forceinline__ v16h load_b_frag(const _Float16* p) {
  return *(const v16h*)p;
}

// ---------------------------------------------------------------------------
// Tiled GEMM: Cout[M,N] = A[M,K] * W[K,N].  A fp32 or f16 (template), W fp32.
// Workgroup: 128 threads (4 waves), 64x64 tile; wave computes 32x32 (2x2 WMMA).
// f16 A-tiles staged via async global->LDS; fp32 converted through VGPRs.
// ---------------------------------------------------------------------------
template <bool A_HALF>
__global__ __launch_bounds__(128) void gemm_wmma_kernel(
    const void* __restrict__ A_, const float* __restrict__ W,
    float* __restrict__ Cout, int M, int N, int K) {
  __shared__ __align__(32) _Float16 As[64 * 32];   // [m][k] row-major
  __shared__ __align__(32) _Float16 Bs[64 * 32];   // transposed: [n][k]

  const int tid  = threadIdx.x;
  const int lane = tid & 31;
  const int wv   = tid >> 5;
  const int ln   = lane & 15;
  const int lh   = lane >> 4;

  const int m0 = blockIdx.y * 64;
  const int n0 = blockIdx.x * 64;
  const int wm = (wv & 1) * 32;
  const int wn = (wv >> 1) * 32;

  v8f acc[2][2] = {};

  for (int k0 = 0; k0 < K; k0 += 32) {
    // ---- stage A tile (64x32) into LDS as f16 ----
    {
      const int ar = tid >> 1;             // 0..63
      const int ac = (tid & 1) * 16;       // 0 or 16
      if constexpr (A_HALF) {
        const _Float16* src =
            (const _Float16*)A_ + (size_t)(m0 + ar) * K + k0 + ac;
        const unsigned dst = lds_off32(&As[ar * 32 + ac]);
        async_copy_b128(dst, src);          // 16B
        async_copy_b128(dst + 16, src + 8); // 16B
      } else {
        const float* src = (const float*)A_ + (size_t)(m0 + ar) * K + k0 + ac;
        const float4* s4 = (const float4*)src;
        float4 f0 = s4[0], f1 = s4[1], f2 = s4[2], f3 = s4[3];
        v8h h0, h1;
        h0[0]=(_Float16)f0.x; h0[1]=(_Float16)f0.y; h0[2]=(_Float16)f0.z; h0[3]=(_Float16)f0.w;
        h0[4]=(_Float16)f1.x; h0[5]=(_Float16)f1.y; h0[6]=(_Float16)f1.z; h0[7]=(_Float16)f1.w;
        h1[0]=(_Float16)f2.x; h1[1]=(_Float16)f2.y; h1[2]=(_Float16)f2.z; h1[3]=(_Float16)f2.w;
        h1[4]=(_Float16)f3.x; h1[5]=(_Float16)f3.y; h1[6]=(_Float16)f3.z; h1[7]=(_Float16)f3.w;
        *(v8h*)&As[ar * 32 + ac]     = h0;
        *(v8h*)&As[ar * 32 + ac + 8] = h1;
      }
    }
    // ---- stage W tile (32x64) transposed into LDS: Bs[n][k] ----
    // Each thread owns two consecutive k-rows x 8 columns -> dword stores.
    {
      const int kr = (tid & 15) * 2;       // 0,2,..,30
      const int nc = (tid >> 4) * 8;       // 0,8,..,56
      const float* s0 = W + (size_t)(k0 + kr) * N + n0 + nc;
      const float* s1 = s0 + N;
      const float4* p0 = (const float4*)s0;
      const float4* p1 = (const float4*)s1;
      float4 a0 = p0[0], a1 = p0[1];
      float4 b0 = p1[0], b1 = p1[1];
      float r0[8] = {a0.x, a0.y, a0.z, a0.w, a1.x, a1.y, a1.z, a1.w};
      float r1[8] = {b0.x, b0.y, b0.z, b0.w, b1.x, b1.y, b1.z, b1.w};
#pragma unroll
      for (int i = 0; i < 8; ++i) {
        v2h pk; pk[0] = (_Float16)r0[i]; pk[1] = (_Float16)r1[i];
        *(v2h*)&Bs[(nc + i) * 32 + kr] = pk;   // 4B store, kr even
      }
    }
    if constexpr (A_HALF) wait_async0();
    __syncthreads();

    v16h af[2], bf[2];
#pragma unroll
    for (int mt = 0; mt < 2; ++mt)
      af[mt] = load_a_frag(&As[(wm + mt * 16 + ln) * 32 + lh * 8]);
#pragma unroll
    for (int nt = 0; nt < 2; ++nt)
      bf[nt] = load_b_frag(&Bs[(wn + nt * 16 + ln) * 32 + lh * 16]);

#pragma unroll
    for (int mt = 0; mt < 2; ++mt)
#pragma unroll
      for (int nt = 0; nt < 2; ++nt)
        acc[mt][nt] = wmma_f16(af[mt], bf[nt], acc[mt][nt]);

    __syncthreads();
  }

  // ---- write 32x32 fp32 result (C/D layout: row = r + lh*8, col = ln) ----
#pragma unroll
  for (int mt = 0; mt < 2; ++mt)
#pragma unroll
    for (int nt = 0; nt < 2; ++nt)
#pragma unroll
      for (int r = 0; r < 8; ++r) {
        const int m = m0 + wm + mt * 16 + r + lh * 8;
        const int n = n0 + wn + nt * 16 + ln;
        Cout[(size_t)m * N + n] = acc[mt][nt][r];
      }
}

// ---------------------------------------------------------------------------
// RoPE + layout transform:
//   qf [B,T,C] fp32      -> q16 [B,H,T,D] f16 (post-RoPE)
//   kf [B,T,Ckv] fp32    -> present_k [B,Hkv,T,D] fp32 + k16 f16 (post-RoPE)
//   vf [B,T,Ckv] fp32    -> present_v [B,Hkv,T,D] fp32 + v16 f16
// One block per (b,t) token row.
// ---------------------------------------------------------------------------
__global__ __launch_bounds__(128) void rope_pack_kernel(
    const float* __restrict__ qf, const float* __restrict__ kf,
    const float* __restrict__ vf, _Float16* __restrict__ q16,
    _Float16* __restrict__ k16, _Float16* __restrict__ v16,
    float* __restrict__ pk, float* __restrict__ pv) {
  const int bt = blockIdx.x;
  const int b  = bt / SEQ_T;
  const int t  = bt % SEQ_T;
  const int tid = threadIdx.x;
  const float ft = (float)t;
  // inv_freq[j] = 10000^(-j/64) = exp(-ln(1e4)/64 * j)
  const float kfreq = -0.14391156514261485f;

  // Q: 16 heads x 64 rotation pairs
  const float* qrow = qf + (size_t)bt * C_DIM;
  for (int i = tid; i < N_HEAD * 64; i += 128) {
    const int h = i >> 6, j = i & 63;
    const float th = __expf(kfreq * (float)j) * ft;
    float s, c;
    __sincosf(th, &s, &c);
    const float x1 = qrow[h * HEAD_D + j];
    const float x2 = qrow[h * HEAD_D + 64 + j];
    const size_t base = (((size_t)(b * N_HEAD + h)) * SEQ_T + t) * HEAD_D;
    q16[base + j]      = (_Float16)(x1 * c - x2 * s);
    q16[base + 64 + j] = (_Float16)(x2 * c + x1 * s);
  }
  // K: 4 kv-heads x 64 pairs (writes fp32 present_k AND f16 k16)
  const float* krow = kf + (size_t)bt * CKV_DIM;
  for (int i = tid; i < N_KVH * 64; i += 128) {
    const int h = i >> 6, j = i & 63;
    const float th = __expf(kfreq * (float)j) * ft;
    float s, c;
    __sincosf(th, &s, &c);
    const float x1 = krow[h * HEAD_D + j];
    const float x2 = krow[h * HEAD_D + 64 + j];
    const float r1 = x1 * c - x2 * s;
    const float r2 = x2 * c + x1 * s;
    const size_t base = (((size_t)(b * N_KVH + h)) * SEQ_T + t) * HEAD_D;
    pk[base + j]       = r1;
    pk[base + 64 + j]  = r2;
    k16[base + j]      = (_Float16)r1;
    k16[base + 64 + j] = (_Float16)r2;
  }
  // V: plain transpose [B,T,Hkv,D] -> [B,Hkv,T,D]
  const float* vrow = vf + (size_t)bt * CKV_DIM;
  for (int i = tid; i < CKV_DIM; i += 128) {
    const int h = i >> 7, d = i & 127;
    const size_t base = (((size_t)(b * N_KVH + h)) * SEQ_T + t) * HEAD_D + d;
    const float val = vrow[i];
    pv[base]  = val;
    v16[base] = (_Float16)val;
  }
}

// ---------------------------------------------------------------------------
// Flash attention. Grid: (T/64, B*H). 128 threads = 4 waves.
// Wave w owns 16 query rows; streams 32-key blocks with online softmax.
// K tile async-copied to LDS (shared by all 4 waves); V tile transposed in LDS.
// ---------------------------------------------------------------------------
__global__ __launch_bounds__(128) void attn_kernel(
    const _Float16* __restrict__ q16, const _Float16* __restrict__ k16,
    const _Float16* __restrict__ v16, _Float16* __restrict__ yatt) {
  __shared__ __align__(32) _Float16 Ks[32 * HEAD_D];   // K tile [k][d], 8KB
  __shared__ __align__(32) _Float16 vT[HEAD_D * 32];   // V tile transposed [d][k]
  __shared__ __align__(32) _Float16 Ps[4][16 * 32];    // per-wave P staging [m][k]

  const int qt = blockIdx.x;
  const int bh = blockIdx.y;
  const int b   = bh / N_HEAD;
  const int h   = bh % N_HEAD;
  const int hkv = h / (N_HEAD / N_KVH);

  const int tid  = threadIdx.x;
  const int lane = tid & 31;
  const int wv   = tid >> 5;
  const int ln   = lane & 15;
  const int lh   = lane >> 4;

  const _Float16* qbase = q16 + ((size_t)(b * N_HEAD + h)) * SEQ_T * HEAD_D;
  const _Float16* kbase = k16 + ((size_t)(b * N_KVH + hkv)) * SEQ_T * HEAD_D;
  const _Float16* vbase = v16 + ((size_t)(b * N_KVH + hkv)) * SEQ_T * HEAD_D;

  const int qrow0 = qt * 64 + wv * 16;

  // Q fragments: 4 D-chunks of 32, kept in registers for the whole stream.
  v16h qfrag[4];
#pragma unroll
  for (int dc = 0; dc < 4; ++dc) {
    const _Float16* p = qbase + (size_t)(qrow0 + ln) * HEAD_D + dc * 32 + lh * 8;
    qfrag[dc] = load_a_frag(p);
  }

  float m_i[8], l_i[8];
#pragma unroll
  for (int r = 0; r < 8; ++r) { m_i[r] = -1e30f; l_i[r] = 0.0f; }
  v8f o[8] = {};                       // 8 d-tiles of 16 columns

  const float scale = 0.08838834764831845f;   // 1/sqrt(128)
  const int nkb = 2 * qt + 2;                 // 32-key blocks covering causal span
  const unsigned ks_base = lds_off32(&Ks[0]);

  for (int kb = 0; kb < nkb; ++kb) {
    const int kstart = kb * 32;

    // ---- async copy K tile (32 keys x 128 d = 8KB, contiguous) to LDS ----
    {
      const char* gsrc = (const char*)(kbase + (size_t)kstart * HEAD_D);
#pragma unroll
      for (int i = 0; i < 4; ++i) {
        const unsigned idx = (unsigned)(i * 128 + tid) * 16u;  // byte chunk
        async_copy_b128(ks_base + idx, gsrc + idx);
      }
    }
    // ---- cooperative load of V tile, transposed into LDS: vT[d][k] ----
    // Each thread owns two consecutive key rows x 16 d-columns -> dword stores.
    {
      const int kr = (tid & 15) * 2;      // 0,2,..,30
      const int c0 = (tid >> 4) * 16;     // 0,16,..,112
      const _Float16* s0 = vbase + (size_t)(kstart + kr) * HEAD_D + c0;
      const _Float16* s1 = s0 + HEAD_D;
      v8h a0 = *(const v8h*)s0, a1 = *(const v8h*)(s0 + 8);
      v8h b0 = *(const v8h*)s1, b1 = *(const v8h*)(s1 + 8);
#pragma unroll
      for (int i = 0; i < 8; ++i) {
        v2h p0; p0[0] = a0[i]; p0[1] = b0[i];
        v2h p1; p1[0] = a1[i]; p1[1] = b1[i];
        *(v2h*)&vT[(c0 + i) * 32 + kr]     = p0;  // 4B store, kr even
        *(v2h*)&vT[(c0 + 8 + i) * 32 + kr] = p1;
      }
    }
    wait_async0();
    __syncthreads();

    // ---- S = Q * K^T : two 16-key tiles, contraction over D (4 WMMAs each) ----
    v8f s[2] = {};
#pragma unroll
    for (int j = 0; j < 2; ++j) {
#pragma unroll
      for (int dc = 0; dc < 4; ++dc) {
        // B fragment from LDS K tile: K-contig halves at lane's key row.
        const _Float16* p = &Ks[(j * 16 + ln) * HEAD_D + dc * 32 + lh * 16];
        s[j] = wmma_f16(qfrag[dc], load_b_frag(p), s[j]);
      }
    }

    // ---- scale + causal mask (row = r + lh*8, col = ln within tile) ----
#pragma unroll
    for (int j = 0; j < 2; ++j)
#pragma unroll
      for (int r = 0; r < 8; ++r) {
        const int qg = qrow0 + r + lh * 8;
        const int kg = kstart + j * 16 + ln;
        s[j][r] = (kg <= qg) ? s[j][r] * scale : -1e30f;
      }

    // ---- online softmax: reductions across the 16-lane half ----
    float mnew[8];
#pragma unroll
    for (int r = 0; r < 8; ++r) {
      float v = fmaxf(s[0][r], s[1][r]);
#pragma unroll
      for (int off = 1; off < 16; off <<= 1)
        v = fmaxf(v, __shfl_xor(v, off, 32));
      mnew[r] = fmaxf(m_i[r], v);
    }
#pragma unroll
    for (int r = 0; r < 8; ++r) {
      s[0][r] = __expf(s[0][r] - mnew[r]);
      s[1][r] = __expf(s[1][r] - mnew[r]);
      float v = s[0][r] + s[1][r];
#pragma unroll
      for (int off = 1; off < 16; off <<= 1)
        v += __shfl_xor(v, off, 32);
      const float alpha = __expf(m_i[r] - mnew[r]);
      l_i[r] = l_i[r] * alpha + v;
      m_i[r] = mnew[r];
#pragma unroll
      for (int dt = 0; dt < 8; ++dt) o[dt][r] *= alpha;
    }

    // ---- stage P (f16) into per-wave LDS, reload in A-fragment layout ----
    _Float16* pw = &Ps[wv][0];
#pragma unroll
    for (int j = 0; j < 2; ++j)
#pragma unroll
      for (int r = 0; r < 8; ++r)
        pw[(r + lh * 8) * 32 + j * 16 + ln] = (_Float16)s[j][r];

    const v16h pfrag = load_a_frag(&pw[ln * 32 + lh * 8]);

    // ---- O += P * V : 8 d-tiles, one K=32 WMMA each ----
#pragma unroll
    for (int dt = 0; dt < 8; ++dt) {
      const _Float16* bp = &vT[(dt * 16 + ln) * 32 + lh * 16];
      o[dt] = wmma_f16(pfrag, load_b_frag(bp), o[dt]);
    }
    __syncthreads();
  }

  // ---- normalize and write attention output in [B,T,H*D] f16 (A for Wo) ----
#pragma unroll
  for (int r = 0; r < 8; ++r) {
    const float inv = 1.0f / l_i[r];
    const int trow = qrow0 + r + lh * 8;
#pragma unroll
    for (int dt = 0; dt < 8; ++dt)
      yatt[((size_t)b * SEQ_T + trow) * C_DIM + h * HEAD_D + dt * 16 + ln] =
          (_Float16)(o[dt][r] * inv);
  }
}

// ---------------------------------------------------------------------------
// Host-side launch.
// ---------------------------------------------------------------------------
extern "C" void kernel_launch(void* const* d_in, const int* in_sizes, int n_in,
                              void* d_out, int out_size, void* d_ws,
                              size_t ws_size, hipStream_t stream) {
  (void)in_sizes; (void)n_in; (void)out_size; (void)ws_size;
  const float* x  = (const float*)d_in[0];
  const float* Wq = (const float*)d_in[1];
  const float* Wk = (const float*)d_in[2];
  const float* Wv = (const float*)d_in[3];
  const float* Wo = (const float*)d_in[4];

  const size_t BTC  = (size_t)M_ROWS * C_DIM;    // 8,388,608
  const size_t BTKV = (size_t)M_ROWS * CKV_DIM;  // 2,097,152

  // Workspace layout (~92 MB)
  char* w = (char*)d_ws;
  float*    qf   = (float*)w;     w += BTC  * sizeof(float);
  float*    kf   = (float*)w;     w += BTKV * sizeof(float);
  float*    vf   = (float*)w;     w += BTKV * sizeof(float);
  _Float16* q16  = (_Float16*)w;  w += BTC  * sizeof(_Float16);
  _Float16* k16  = (_Float16*)w;  w += BTKV * sizeof(_Float16);
  _Float16* v16  = (_Float16*)w;  w += BTKV * sizeof(_Float16);
  _Float16* yatt = (_Float16*)w;  w += BTC  * sizeof(_Float16);

  float* y  = (float*)d_out;          // [B,T,C]
  float* pk = y + BTC;                // present_k [B,Hkv,T,D]
  float* pv = pk + BTKV;              // present_v [B,Hkv,T,D]

  const dim3 blk(128);

  // QKV projections (fp32 A path)
  gemm_wmma_kernel<false><<<dim3(C_DIM / 64, M_ROWS / 64), blk, 0, stream>>>(
      x, Wq, qf, M_ROWS, C_DIM, C_DIM);
  gemm_wmma_kernel<false><<<dim3(CKV_DIM / 64, M_ROWS / 64), blk, 0, stream>>>(
      x, Wk, kf, M_ROWS, CKV_DIM, C_DIM);
  gemm_wmma_kernel<false><<<dim3(CKV_DIM / 64, M_ROWS / 64), blk, 0, stream>>>(
      x, Wv, vf, M_ROWS, CKV_DIM, C_DIM);

  // RoPE + transpose + dtype packing (also emits present_k / present_v)
  rope_pack_kernel<<<dim3(BATCH * SEQ_T), blk, 0, stream>>>(
      qf, kf, vf, q16, k16, v16, pk, pv);

  // Flash attention
  attn_kernel<<<dim3(SEQ_T / 64, BATCH * N_HEAD), blk, 0, stream>>>(
      q16, k16, v16, yatt);

  // Output projection (f16 A path)
  gemm_wmma_kernel<true><<<dim3(C_DIM / 64, M_ROWS / 64), blk, 0, stream>>>(
      yatt, Wo, y, M_ROWS, C_DIM, C_DIM);
}
